// TFTDictBlock_60395830116631
// MI455X (gfx1250) — matrix operand order repaired and checked by reading, not compiled, not run on hardware
//
#include <hip/hip_runtime.h>
#include <hip/hip_bf16.h>
#include <math.h>

// ---------------------------------------------------------------------------
// Problem constants (reference: B=1, T=1024, D=768, H=12, FF=3072, V=8192)
// ---------------------------------------------------------------------------
#define T_  1024
#define D_  768
#define H_  12
#define HD_ 64
#define FF_ 3072
#define V_  8192

typedef unsigned short u16;
typedef unsigned int   u32;
typedef __attribute__((ext_vector_type(16))) __bf16 v16bf;
typedef __attribute__((ext_vector_type(8)))  float  v8f;
typedef int v4i __attribute__((vector_size(4 * sizeof(int))));   // matches builtin param

union Frag  { v16bf v; u32 u[8]; };
union CFrag { v8f   v; float f[8]; };

__device__ __forceinline__ u16 f32_to_bf16(float f) {
    u32 u = __float_as_uint(f);
    u32 r = u + 0x7FFFu + ((u >> 16) & 1u);   // round-to-nearest-even
    return (u16)(r >> 16);
}

// gfx1250 async global->LDS copy (ASYNCcnt-tracked), if the toolchain exposes it
#if __has_builtin(__builtin_amdgcn_global_load_async_to_lds_b128) && \
    __has_builtin(__builtin_amdgcn_s_wait_asynccnt)
#define USE_ASYNC 1
// global (addrspace 1) / LDS (addrspace 3) v4i pointers for the builtin
#define GV4I(p) ((__attribute__((address_space(1))) v4i*)(v4i*)(p))
#define LV4I(p) ((__attribute__((address_space(3))) v4i*)(v4i*)(p))
#else
#define USE_ASYNC 0
#endif

enum { EPI_F32 = 0, EPI_BIAS_F32 = 1, EPI_GELU_BF16 = 2, EPI_QK = 3 };

template <int EPI>
__device__ __forceinline__ void epi_store(float v, int m, int n, int z,
                                          float* Cf, u16* Cu, int ldc, long long sCz,
                                          const float* bias, long long sBiasZ,
                                          u16* qbuf, u16* kbuf)
{
    if (EPI == EPI_F32) {
        Cf[(size_t)z * sCz + (size_t)m * ldc + n] = v;
    } else if (EPI == EPI_BIAS_F32) {
        v += bias[(size_t)z * sBiasZ + n];
        Cf[(size_t)z * sCz + (size_t)m * ldc + n] = v;
    } else if (EPI == EPI_GELU_BF16) {
        v += bias[(size_t)z * sBiasZ + n];
        v = 0.5f * v * (1.0f + erff(v * 0.70710678118f));   // exact gelu
        Cu[(size_t)z * sCz + (size_t)m * ldc + n] = f32_to_bf16(v);
    } else { // EPI_QK: scatter q (pre-scaled by 1/sqrt(HD)) and k to [H,T,HD] bf16
        v += bias[n];
        if (n < D_) {
            int hh = n >> 6, d = n & 63;
            qbuf[((size_t)hh * T_ + m) * HD_ + d] = f32_to_bf16(v * 0.125f);
        } else {
            int n2 = n - D_; int hh = n2 >> 6, d = n2 & 63;
            kbuf[((size_t)hh * T_ + m) * HD_ + d] = f32_to_bf16(v);
        }
    }
}

// Stage one 128xTN K=32 bf16 tile pair into an LDS buffer.
// Async path: GLOBAL_LOAD_ASYNC_TO_LDS_B128 per 16B chunk (no VGPR data, no
// spill risk, overlap via ASYNCcnt). Sync fallback: immediate load+ds_store
// (no cross-iteration register liveness).
template <int TN>
__device__ __forceinline__ void stage_tile(const u32* __restrict__ A32,
                                           const u32* __restrict__ B32,
                                           int ldaU, int ldbU, int mBase, int nBase,
                                           int kU, u32* a, u32* b, int tid)
{
    constexpr int RB = (TN * 4) / 256;
#if USE_ASYNC
    #pragma unroll
    for (int i = 0; i < 2; ++i) {
        int id = tid + i * 256, r = id >> 2, c = id & 3;
        const u32* gp = A32 + (size_t)(mBase + r) * ldaU + kU + c * 4;
        __builtin_amdgcn_global_load_async_to_lds_b128(
            GV4I((u32*)gp), LV4I(&a[r * 20 + c * 4]), 0, 0);
    }
    #pragma unroll
    for (int i = 0; i < RB; ++i) {
        int id = tid + i * 256, r = id >> 2, c = id & 3;
        const u32* gp = B32 + (size_t)(nBase + r) * ldbU + kU + c * 4;
        __builtin_amdgcn_global_load_async_to_lds_b128(
            GV4I((u32*)gp), LV4I(&b[r * 20 + c * 4]), 0, 0);
    }
#else
    #pragma unroll
    for (int i = 0; i < 2; ++i) {
        int id = tid + i * 256, r = id >> 2, c = id & 3;
        uint4 t = ((const uint4*)(A32 + (size_t)(mBase + r) * ldaU + kU))[c];
        *(uint4*)&a[r * 20 + c * 4] = t;
    }
    #pragma unroll
    for (int i = 0; i < RB; ++i) {
        int id = tid + i * 256, r = id >> 2, c = id & 3;
        uint4 t = ((const uint4*)(B32 + (size_t)(nBase + r) * ldbU + kU))[c];
        *(uint4*)&b[r * 20 + c * 4] = t;
    }
#endif
}

// ---------------------------------------------------------------------------
// WMMA bf16 GEMM:  C[m,n] = sum_k A[m,k] * B[n,k]  ("NT", both row-major;
// B fragments use the ISA's 32x16 K-major column layout).
//
// Tile: 128 x TN (TN = 128 or 64) per 256-thread block (8 waves).
//   TN=128: wave grid 4x2, each wave 32x64 -> 2 A-frags x 4 B-frags = 8 wmma/step
//   TN=64 : wave grid 8x1, each wave 16x64 -> 1 A-frag x 4 B-frags = 4 wmma/step
// Double-buffered LDS; tile k+1 streams in via async-to-LDS while WMMAs run
// on tile k (s_wait_asynccnt keeps only the current tile's loads blocking).
// LDS pitch = 20 uints: 16B-aligned chunks (ds_load_b128) and 16 distinct
// banks across the 16-lane fragment reads. Packing per CDNA5 ISA 7.12.2.
// ---------------------------------------------------------------------------
template <int EPI, int TN>
__global__ __launch_bounds__(256)
void gemm_nt_bf16(const u16* __restrict__ Ab, int lda, long long sAz,
                  const u16* __restrict__ Bb, int ldb, long long sBz,
                  float* __restrict__ Cf, u16* __restrict__ Cu, int ldc, long long sCz,
                  const float* __restrict__ bias, long long sBiasZ,
                  int K, int causal,
                  u16* __restrict__ qbuf, u16* __restrict__ kbuf)
{
    constexpr int TM  = 128;
    constexpr int MF  = (TN == 128) ? 2 : 1;    // A fragments per wave
    constexpr int NF  = 4;                      // B fragments per wave
    constexpr int P   = 20;                     // LDS pitch (uints)
    constexpr int NPT = 2 + (TN * 4) / 256;     // async issues per thread per tile

    const int z     = blockIdx.z;
    const int mBase = blockIdx.y * TM;
    const int nBase = blockIdx.x * TN;
    if (causal && nBase > mBase + TM - 1) return;   // tile fully above diagonal

    const u32* A32 = (const u32*)(Ab + (size_t)z * sAz);
    const u32* B32 = (const u32*)(Bb + (size_t)z * sBz);
    const int ldaU = lda >> 1, ldbU = ldb >> 1;

    __shared__ u32 lA[2 * TM * P];
    __shared__ u32 lB[2 * TN * P];

    const int tid  = threadIdx.x;
    const int lane = tid & 31;
    const int wave = tid >> 5;
    const int hsel = lane >> 4;       // 0/1 -> K-half selector
    const int lmod = lane & 15;

    int mSub, nSub;
    if (TN == 128) { mSub = (wave & 3) * 32; nSub = (wave >> 2) * 64; }
    else           { mSub = wave * 16;       nSub = 0; }

    CFrag acc[MF][NF];
    #pragma unroll
    for (int x = 0; x < MF; ++x)
        #pragma unroll
        for (int y = 0; y < NF; ++y)
            #pragma unroll
            for (int i = 0; i < 8; ++i) acc[x][y].f[i] = 0.0f;

    // prologue: start tile k0=0 into buffer 0
    stage_tile<TN>(A32, B32, ldaU, ldbU, mBase, nBase, 0, lA, lB, tid);

    int buf = 0;
    for (int k0 = 0; k0 < K; k0 += 32) {
        // stream next tile into the other buffer while this one is consumed
        if (k0 + 32 < K) {
            stage_tile<TN>(A32, B32, ldaU, ldbU, mBase, nBase, (k0 + 32) >> 1,
                           lA + (buf ^ 1) * TM * P, lB + (buf ^ 1) * TN * P, tid);
#if USE_ASYNC
            __builtin_amdgcn_s_wait_asynccnt(NPT);   // tile k landed; k+1 in flight
#endif
        } else {
#if USE_ASYNC
            __builtin_amdgcn_s_wait_asynccnt(0);
#endif
        }
        if (k0 + 64 < K) {   // gfx1250 global_prefetch_b8, two tiles ahead into L2
            __builtin_prefetch(A32 + (size_t)(mBase + (tid >> 1)) * ldaU + ((k0 + 64) >> 1), 0, 0);
        }
        __syncthreads();

        const u32* aB = lA + buf * TM * P;
        const u32* bB = lB + buf * TN * P;

        // A fragments (16x32 bf16): lanes 0-15 K={0..7,16..23}, lanes 16-31 K={8..15,24..31}
        Frag afr[MF];
        #pragma unroll
        for (int x = 0; x < MF; ++x) {
            const u32* row = &aB[(mSub + x * 16 + lmod) * P];
            #pragma unroll
            for (int i = 0; i < 4; ++i) afr[x].u[i]     = row[hsel * 4 + i];
            #pragma unroll
            for (int i = 0; i < 4; ++i) afr[x].u[4 + i] = row[8 + hsel * 4 + i];
        }
        #pragma unroll
        for (int y = 0; y < NF; ++y) {
            // B fragment (32x16 bf16): lanes 0-15 K={0..7,16..23}, lanes 16-31 K={8..15,24..31}
            Frag bfr;
            const u32* row = &bB[(nSub + y * 16 + lmod) * P];
            #pragma unroll
            for (int i = 0; i < 8; ++i) bfr.u[i] = row[(i >> 2) * 8 + hsel * 4 + (i & 3)];
            #pragma unroll
            for (int x = 0; x < MF; ++x)
                acc[x][y].v = __builtin_amdgcn_wmma_f32_16x16x32_bf16(
                    false, afr[x].v, false, bfr.v, (short)0, acc[x][y].v, false, false);
        }
        __syncthreads();   // all reads of buf done before it is refilled at k+2
        buf ^= 1;
    }

    // Epilogue. C/D layout: lane 0-15 -> N=lane, rows M=0..7 in f[0..7];
    // lane 16-31 -> N=lane-16, rows M=8..15.
    #pragma unroll
    for (int x = 0; x < MF; ++x) {
        const int mRow0 = mBase + mSub + x * 16 + hsel * 8;
        #pragma unroll
        for (int y = 0; y < NF; ++y) {
            const int n = nBase + nSub + y * 16 + lmod;
            #pragma unroll
            for (int r = 0; r < 8; ++r)
                epi_store<EPI>(acc[x][y].f[r], mRow0 + r, n, z,
                               Cf, Cu, ldc, sCz, bias, sBiasZ, qbuf, kbuf);
        }
    }
}

// ---------------------------------------------------------------------------
// Elementwise / reduction kernels
// ---------------------------------------------------------------------------
__global__ void cvt_bf16_kernel(const float* __restrict__ in, u16* __restrict__ out, long long n) {
    long long i = (long long)blockIdx.x * blockDim.x + threadIdx.x;
    long long step = (long long)gridDim.x * blockDim.x;
    for (; i < n; i += step) out[i] = f32_to_bf16(in[i]);
}

// out = bf16( LayerNorm(a+b) * w + bb ), one 768-wide row per block
__global__ __launch_bounds__(256)
void add_ln_bf16_kernel(const float* __restrict__ a, const float* __restrict__ b,
                        const float* __restrict__ w, const float* __restrict__ bb,
                        u16* __restrict__ out)
{
    __shared__ float red[256];
    const int t = blockIdx.x, tid = threadIdx.x;
    const float* ar = a + (size_t)t * D_;
    const float* br = b + (size_t)t * D_;
    float x[3], s = 0.f, s2 = 0.f;
    #pragma unroll
    for (int p = 0; p < 3; ++p) {
        int c = tid + p * 256;
        x[p] = ar[c] + br[c]; s += x[p]; s2 += x[p] * x[p];
    }
    red[tid] = s; __syncthreads();
    for (int q = 128; q > 0; q >>= 1) { if (tid < q) red[tid] += red[tid + q]; __syncthreads(); }
    s = red[0]; __syncthreads();
    red[tid] = s2; __syncthreads();
    for (int q = 128; q > 0; q >>= 1) { if (tid < q) red[tid] += red[tid + q]; __syncthreads(); }
    s2 = red[0];
    const float mu  = s * (1.0f / D_);
    const float var = s2 * (1.0f / D_) - mu * mu;
    const float inv = rsqrtf(var + 1e-5f);
    u16* orow = out + (size_t)t * D_;
    #pragma unroll
    for (int p = 0; p < 3; ++p) {
        int c = tid + p * 256;
        orow[c] = f32_to_bf16((x[p] - mu) * inv * w[c] + bb[c]);
    }
}

// v_t[h][d][t] = bf16( sum_j v_fact[h,j] * xt[t, j*64+d] )   (stored transposed)
__global__ void v_mix_kernel(const float* __restrict__ xt, const float* __restrict__ vfact,
                             u16* __restrict__ vt)
{
    int idx = blockIdx.x * blockDim.x + threadIdx.x;     // H*HD*T
    if (idx >= H_ * HD_ * T_) return;
    int t = idx & (T_ - 1);
    int d = (idx >> 10) & 63;
    int h = idx >> 16;
    float acc = 0.f;
    #pragma unroll
    for (int j = 0; j < H_; ++j) acc += vfact[h * H_ + j] * xt[(size_t)t * D_ + j * HD_ + d];
    vt[((size_t)h * HD_ + d) * T_ + t] = f32_to_bf16(acc);
}

// softmax over j with alibi bias + causal mask; scores f32 -> attn bf16
__global__ __launch_bounds__(256)
void attn_softmax_kernel(const float* __restrict__ scores, u16* __restrict__ attn)
{
    __shared__ float red[256];
    const int h = blockIdx.x >> 10;
    const int i = blockIdx.x & (T_ - 1);
    const int tid = threadIdx.x;
    const float slope = (h < 8) ? exp2f(-(float)(h + 1)) : exp2f(-0.5f * (float)(h - 7));
    const float* srow = scores + ((size_t)h * T_ + i) * T_;
    float v[4];
    #pragma unroll
    for (int p = 0; p < 4; ++p) {
        int j = tid + p * 256;
        v[p] = (j <= i) ? (srow[j] + slope * (float)(j - i)) : -INFINITY;
    }
    float mx = fmaxf(fmaxf(v[0], v[1]), fmaxf(v[2], v[3]));
    red[tid] = mx; __syncthreads();
    for (int q = 128; q > 0; q >>= 1) { if (tid < q) red[tid] = fmaxf(red[tid], red[tid + q]); __syncthreads(); }
    mx = red[0]; __syncthreads();
    float e[4], s = 0.f;
    #pragma unroll
    for (int p = 0; p < 4; ++p) { e[p] = (v[p] == -INFINITY) ? 0.f : expf(v[p] - mx); s += e[p]; }
    red[tid] = s; __syncthreads();
    for (int q = 128; q > 0; q >>= 1) { if (tid < q) red[tid] += red[tid + q]; __syncthreads(); }
    const float inv = 1.0f / red[0];
    u16* arow = attn + ((size_t)h * T_ + i) * T_;
    #pragma unroll
    for (int p = 0; p < 4; ++p) arow[tid + p * 256] = f32_to_bf16(e[p] * inv);
}

// xt_new[t, i*64+d] = xt + sum_j out_fact[i,j] * y_heads[j][t][d]
__global__ void out_mix_kernel(const float* __restrict__ xt, const float* __restrict__ ofact,
                               const float* __restrict__ yheads, float* __restrict__ xtnew)
{
    int idx = blockIdx.x * blockDim.x + threadIdx.x;     // T*D
    if (idx >= T_ * D_) return;
    int t = idx / D_, c = idx % D_, i = c >> 6, d = c & 63;
    float acc = xt[idx];
    #pragma unroll
    for (int j = 0; j < H_; ++j) acc += ofact[i * H_ + j] * yheads[((size_t)j * T_ + t) * HD_ + d];
    xtnew[idx] = acc;
}

// per-head 64-wide LayerNorm -> xhat f32 + bf16
__global__ __launch_bounds__(64)
void dict_ln_kernel(const float* __restrict__ pre, const float* __restrict__ w,
                    const float* __restrict__ b, float* __restrict__ xhat,
                    u16* __restrict__ xhatb)
{
    __shared__ float red[64];
    const int row = blockIdx.x;          // h*T + t
    const int h = row >> 10;
    const int e = threadIdx.x;
    const float x = pre[(size_t)row * HD_ + e];
    red[e] = x; __syncthreads();
    for (int q = 32; q > 0; q >>= 1) { if (e < q) red[e] += red[e + q]; __syncthreads(); }
    const float mu = red[0] * (1.0f / HD_); __syncthreads();
    const float dd = x - mu;
    red[e] = dd * dd; __syncthreads();
    for (int q = 32; q > 0; q >>= 1) { if (e < q) red[e] += red[e + q]; __syncthreads(); }
    const float inv = rsqrtf(red[0] * (1.0f / HD_) + 1e-5f);
    const float o = dd * inv * w[h * HD_ + e] + b[h * HD_ + e];
    xhat[(size_t)row * HD_ + e]  = o;
    xhatb[(size_t)row * HD_ + e] = f32_to_bf16(o);
}

// softmax over V=8192 (in-place on d_out dict_weights) + x_recon + xe_new + loss
__global__ __launch_bounds__(256)
void dict_softmax_recon_kernel(float* __restrict__ logits_base,        // d_out + DW
                               const float* __restrict__ emb,          // f32 dict_emb
                               const float* __restrict__ xhat,
                               const float* __restrict__ xe,
                               float* __restrict__ xe_new,
                               float* __restrict__ loss)
{
    __shared__ float red[256];
    const int h = blockIdx.x >> 10;
    const int t = blockIdx.x & (T_ - 1);
    const int tid = threadIdx.x;
    float* lrow = logits_base + ((size_t)h * T_ + t) * V_;
    const float* embh = emb + (size_t)h * V_ * HD_;

    float mx = -INFINITY;
    for (int v = tid; v < V_; v += 256) mx = fmaxf(mx, lrow[v]);
    red[tid] = mx; __syncthreads();
    for (int q = 128; q > 0; q >>= 1) { if (tid < q) red[tid] = fmaxf(red[tid], red[tid + q]); __syncthreads(); }
    mx = red[0]; __syncthreads();

    float s = 0.f;
    for (int v = tid; v < V_; v += 256) s += expf(lrow[v] - mx);
    red[tid] = s; __syncthreads();
    for (int q = 128; q > 0; q >>= 1) { if (tid < q) red[tid] += red[tid + q]; __syncthreads(); }
    const float inv = 1.0f / red[0]; __syncthreads();

    // recon: threads = 4 v-groups x 64 e-lanes, reading raw logits
    const int e = tid & 63, g = tid >> 6;
    float acc = 0.f;
    for (int v = g; v < V_; v += 4) {
        const float wgt = expf(lrow[v] - mx) * inv;
        acc += wgt * embh[(size_t)v * HD_ + e];
    }
    __syncthreads();
    // write normalized weights in place
    for (int v = tid; v < V_; v += 256) lrow[v] = expf(lrow[v] - mx) * inv;

    red[tid] = acc; __syncthreads();
    float diff2 = 0.f;
    if (g == 0) {
        const float r = red[e] + red[64 + e] + red[128 + e] + red[192 + e];
        const size_t off = (size_t)t * D_ + h * HD_ + e;
        xe_new[off] = xe[off] + r;
        const float d = xhat[((size_t)h * T_ + t) * HD_ + e] - r;
        diff2 = d * d;
    }
    __syncthreads();
    red[tid] = diff2; __syncthreads();
    for (int q = 128; q > 0; q >>= 1) { if (tid < q) red[tid] += red[tid + q]; __syncthreads(); }
    if (tid == 0) atomicAdd(loss, red[0] * (1.0f / (float)(H_ * T_ * HD_)));
}

// ---------------------------------------------------------------------------
// Host-side orchestration
// ---------------------------------------------------------------------------
extern "C" void kernel_launch(void* const* d_in, const int* in_sizes, int n_in,
                              void* d_out, int out_size, void* d_ws, size_t ws_size,
                              hipStream_t stream)
{
    const float* xt      = (const float*)d_in[0];
    const float* xe      = (const float*)d_in[1];
    const float* ln1_w   = (const float*)d_in[2];
    const float* ln1_b   = (const float*)d_in[3];
    const float* qk_w    = (const float*)d_in[4];
    const float* qk_b    = (const float*)d_in[5];
    const float* v_fact  = (const float*)d_in[6];
    const float* out_fact= (const float*)d_in[7];
    const float* ln2_w   = (const float*)d_in[8];
    const float* ln2_b   = (const float*)d_in[9];
    const float* fc_w    = (const float*)d_in[10];
    const float* fc_b    = (const float*)d_in[11];
    const float* proj_w  = (const float*)d_in[12];
    const float* proj_b  = (const float*)d_in[13];
    const float* dln_w   = (const float*)d_in[14];
    const float* dln_b   = (const float*)d_in[15];
    const float* dict_emb= (const float*)d_in[16];

    float* out = (float*)d_out;
    const size_t XT_OFF = 0;
    const size_t XE_OFF = (size_t)T_ * D_;                 // 786432
    const size_t DW_OFF = 2 * (size_t)T_ * D_;             // 1572864
    const size_t LS_OFF = DW_OFF + (size_t)H_ * T_ * V_;   // 102236160

    // workspace allocator (256B aligned)
    size_t cur = 0;
    auto alloc = [&](size_t bytes) -> char* {
        cur = (cur + 255) & ~(size_t)255;
        char* p = (char*)d_ws + cur;
        cur += bytes;
        return p;
    };
    u16*  qkwB   = (u16*) alloc((size_t)2 * D_ * D_ * 2);
    u16*  fcwB   = (u16*) alloc((size_t)H_ * FF_ * D_ * 2);
    u16*  projwB = (u16*) alloc((size_t)H_ * HD_ * FF_ * 2);
    u16*  embB   = (u16*) alloc((size_t)H_ * V_ * HD_ * 2);
    u16*  xnorm  = (u16*) alloc((size_t)T_ * D_ * 2);
    u16*  qbuf   = (u16*) alloc((size_t)H_ * T_ * HD_ * 2);
    u16*  kbuf   = (u16*) alloc((size_t)H_ * T_ * HD_ * 2);
    u16*  vtB    = (u16*) alloc((size_t)H_ * HD_ * T_ * 2);
    char* region = alloc((size_t)H_ * T_ * T_ * 6);        // == H*T*FF*2 (aliased)
    float* scores = (float*)region;                        // [H,T,T] f32
    u16*   attnB  = (u16*)(region + (size_t)H_ * T_ * T_ * 4);
    u16*   hB     = (u16*)region;                          // reuses region after attention
    float* yheads = (float*)alloc((size_t)H_ * T_ * HD_ * 4);
    u16*   x2B    = (u16*) alloc((size_t)T_ * D_ * 2);
    float* xhpre  = (float*)alloc((size_t)H_ * T_ * HD_ * 4);
    float* xhat   = (float*)alloc((size_t)H_ * T_ * HD_ * 4);
    u16*   xhatB  = (u16*) alloc((size_t)H_ * T_ * HD_ * 2);
    (void)ws_size; (void)n_in; (void)in_sizes; (void)out_size;

    (void)hipMemsetAsync(out + LS_OFF, 0, sizeof(float), stream);

    // weight conversions f32 -> bf16
    cvt_bf16_kernel<<<2048, 256, 0, stream>>>(qk_w,    qkwB,   (long long)2 * D_ * D_);
    cvt_bf16_kernel<<<2048, 256, 0, stream>>>(fc_w,    fcwB,   (long long)H_ * FF_ * D_);
    cvt_bf16_kernel<<<2048, 256, 0, stream>>>(proj_w,  projwB, (long long)H_ * HD_ * FF_);
    cvt_bf16_kernel<<<2048, 256, 0, stream>>>(dict_emb,embB,   (long long)H_ * V_ * HD_);

    // ln1(xt+xe) -> x_norm bf16
    add_ln_bf16_kernel<<<T_, 256, 0, stream>>>(xt, xe, ln1_w, ln1_b, xnorm);

    // qk GEMM: [1024,1536] = x_norm @ qk_w^T, scatter to q/k head buffers
    gemm_nt_bf16<EPI_QK, 128><<<dim3(2 * D_ / 128, T_ / 128, 1), 256, 0, stream>>>(
        xnorm, D_, 0, qkwB, D_, 0, nullptr, nullptr, 0, 0, qk_b, 0,
        D_, 0, qbuf, kbuf);

    // v = kron(v_fact) applied to xt heads, stored transposed [H,HD,T]
    v_mix_kernel<<<(H_ * HD_ * T_) / 256, 256, 0, stream>>>(xt, v_fact, vtB);

    // scores[h] = q[h] @ k[h]^T (scale folded into q), causal tiles skipped
    gemm_nt_bf16<EPI_F32, 128><<<dim3(T_ / 128, T_ / 128, H_), 256, 0, stream>>>(
        qbuf, HD_, (long long)T_ * HD_, kbuf, HD_, (long long)T_ * HD_,
        scores, nullptr, T_, (long long)T_ * T_, nullptr, 0,
        HD_, 1, nullptr, nullptr);

    // alibi + causal softmax -> attn bf16
    attn_softmax_kernel<<<H_ * T_, 256, 0, stream>>>(scores, attnB);

    // y_heads[h] = attn[h] @ v[h]  (v stored transposed -> NT form)
    gemm_nt_bf16<EPI_F32, 64><<<dim3(1, T_ / 128, H_), 256, 0, stream>>>(
        attnB, T_, (long long)T_ * T_, vtB, T_, (long long)HD_ * T_,
        yheads, nullptr, HD_, (long long)T_ * HD_, nullptr, 0,
        T_, 0, nullptr, nullptr);

    // xt_new = xt + kron(out_fact) y  -> d_out
    out_mix_kernel<<<(T_ * D_) / 256, 256, 0, stream>>>(xt, out_fact, yheads, out + XT_OFF);

    // ln2(xt_new + xe) -> x2 bf16
    add_ln_bf16_kernel<<<T_, 256, 0, stream>>>(out + XT_OFF, xe, ln2_w, ln2_b, x2B);

    // fc GEMM per head + bias + gelu -> h bf16 (aliases dead attn/scores region)
    gemm_nt_bf16<EPI_GELU_BF16, 128><<<dim3(FF_ / 128, T_ / 128, H_), 256, 0, stream>>>(
        x2B, D_, 0, fcwB, D_, (long long)FF_ * D_,
        nullptr, hB, FF_, (long long)T_ * FF_, fc_b, FF_,
        D_, 0, nullptr, nullptr);

    // proj GEMM per head + bias -> xhat_pre f32
    gemm_nt_bf16<EPI_BIAS_F32, 64><<<dim3(1, T_ / 128, H_), 256, 0, stream>>>(
        hB, FF_, (long long)T_ * FF_, projwB, FF_, (long long)HD_ * FF_,
        xhpre, nullptr, HD_, (long long)T_ * HD_, proj_b, HD_,
        FF_, 0, nullptr, nullptr);

    // per-head 64-wide LN -> xhat (f32 + bf16)
    dict_ln_kernel<<<H_ * T_, 64, 0, stream>>>(xhpre, dln_w, dln_b, xhat, xhatB);

    // dict logits GEMM per head, written directly into d_out dict_weights region
    gemm_nt_bf16<EPI_F32, 128><<<dim3(V_ / 128, T_ / 128, H_), 256, 0, stream>>>(
        xhatB, HD_, (long long)T_ * HD_, embB, HD_, (long long)V_ * HD_,
        out + DW_OFF, nullptr, V_, (long long)T_ * V_, nullptr, 0,
        HD_, 0, nullptr, nullptr);

    // softmax over V (in place) + x_recon + xe_new + dict_loss
    dict_softmax_recon_kernel<<<H_ * T_, 256, 0, stream>>>(
        out + DW_OFF, dict_emb, xhat, xe, out + XE_OFF, out + LS_OFF);
}
